// SNN_46669114638758
// MI455X (gfx1250) — matrix operand order repaired and checked by reading, not compiled, not run on hardware
//
#include <hip/hip_runtime.h>

// Fused SNN forward for MI455X (gfx1250, wave32):
//   fc1 (bf16 hi/lo split x3 WMMA, fp32 accum)  -> cur1 in VGPRs
//   25 LIF steps with mem1 resident in VGPRs, spk1 via LDS (bf16)
//   fc2 per step (spk1 exact-bf16 A, W2 hi/lo split x2 WMMA) by waves 0/1,
//   W2 hi/lo staged in LDS once.
// Outputs: d_out = [spk_rec (25*32768*10) | mem_rec (25*32768*10)] fp32.

typedef __attribute__((ext_vector_type(16))) __bf16 v16bf;
typedef __attribute__((ext_vector_type(8)))  __bf16 v8bf;
typedef __attribute__((ext_vector_type(8)))  float  v8f;

#define DIN   784
#define DH    512
#define DOUT  10
#define BATCH 32768
#define STEPS 25
#define BM    32      // batch rows per block
#define SPAD  520     // LDS row stride in bf16 elems (1040 B = 260 dwords -> +4 banks/row)

// Convert a float4 into bf16 hi + lo parts at constant element base Bo.
#define CVT4(vhi, vlo, F, Bo) do {                                          \
    float _f0=(F).x,_f1=(F).y,_f2=(F).z,_f3=(F).w;                          \
    __bf16 _h0=(__bf16)_f0,_h1=(__bf16)_f1,_h2=(__bf16)_f2,_h3=(__bf16)_f3; \
    (vhi)[(Bo)+0]=_h0; (vhi)[(Bo)+1]=_h1;                                   \
    (vhi)[(Bo)+2]=_h2; (vhi)[(Bo)+3]=_h3;                                   \
    (vlo)[(Bo)+0]=(__bf16)(_f0-(float)_h0);                                 \
    (vlo)[(Bo)+1]=(__bf16)(_f1-(float)_h1);                                 \
    (vlo)[(Bo)+2]=(__bf16)(_f2-(float)_h2);                                 \
    (vlo)[(Bo)+3]=(__bf16)(_f3-(float)_h3);                                 \
} while (0)

#define WMMA_BF16(A, Bm, C) \
    __builtin_amdgcn_wmma_f32_16x16x32_bf16(false,(A),false,(Bm),(short)0,(C),false,false)

// One K-chunk of the fc1 GEMM. TAIL is a literal so branches constant-fold.
#define FC1_CHUNK(KK, TAIL) do {                                              \
    const int kk = (KK);                                                      \
    v16bf ah[2], al[2];                                                       \
    _Pragma("unroll")                                                         \
    for (int mt = 0; mt < 2; ++mt) {                                          \
        const float* xp = x + (size_t)(mbase + mt*16 + lh) * DIN + kk + hh*8; \
        float4 f0 = *(const float4*)(xp);                                     \
        float4 f1 = *(const float4*)(xp + 4);                                 \
        float4 f2, f3;                                                        \
        if (!(TAIL)) {                                                        \
            f2 = *(const float4*)(xp + 16); f3 = *(const float4*)(xp + 20);   \
            __builtin_prefetch(xp + 32, 0, 3);    /* next K chunk */          \
        } else { f2 = make_float4(0.f,0.f,0.f,0.f); f3 = f2; }                \
        CVT4(ah[mt], al[mt], f0, 0);  CVT4(ah[mt], al[mt], f1, 4);            \
        CVT4(ah[mt], al[mt], f2, 8);  CVT4(ah[mt], al[mt], f3, 12);           \
    }                                                                         \
    _Pragma("unroll")                                                         \
    for (int nt = 0; nt < 4; ++nt) {                                          \
        v16bf bh, bl;                                                         \
        const float* wp = W1 + (size_t)(wav*64 + nt*16 + lh) * DIN + kk + hh*8; \
        float4 g0, g1, g2, g3;                                                \
        g0 = *(const float4*)(wp);     g1 = *(const float4*)(wp + 4);         \
        if (!(TAIL)) {                                                        \
            g2 = *(const float4*)(wp + 16); g3 = *(const float4*)(wp + 20);   \
        } else {                                                              \
            g2 = make_float4(0.f,0.f,0.f,0.f); g3 = g2;                       \
        }                                                                     \
        if (!(TAIL)) __builtin_prefetch(wp + 32, 0, 3);                       \
        CVT4(bh, bl, g0, 0);  CVT4(bh, bl, g1, 4);                            \
        CVT4(bh, bl, g2, 8);  CVT4(bh, bl, g3, 12);                           \
        _Pragma("unroll")                                                     \
        for (int mt = 0; mt < 2; ++mt) {                                      \
            cur1[mt][nt] = WMMA_BF16(ah[mt], bh, cur1[mt][nt]);  /* hi*hi */  \
            cur1[mt][nt] = WMMA_BF16(ah[mt], bl, cur1[mt][nt]);  /* hi*lo */  \
            cur1[mt][nt] = WMMA_BF16(al[mt], bh, cur1[mt][nt]);  /* lo*hi */  \
        }                                                                     \
    }                                                                         \
} while (0)

__global__ __launch_bounds__(256) void snn_fused(
    const float* __restrict__ x,  const float* __restrict__ W1,
    const float* __restrict__ b1, const float* __restrict__ W2,
    const float* __restrict__ b2, float* __restrict__ out)
{
    __shared__ __align__(16) __bf16 s_spk[BM * SPAD];    // 33,280 B
    __shared__ __align__(16) __bf16 s_w2h[DOUT * SPAD];  // 10,400 B
    __shared__ __align__(16) __bf16 s_w2l[DOUT * SPAD];  // 10,400 B

    const int tid   = threadIdx.x;
    const int lane  = tid & 31;
    const int wav   = tid >> 5;     // 8 waves
    const int lh    = lane & 15;    // lane within half-wave
    const int hh    = lane >> 4;    // half-wave index
    const int mbase = blockIdx.x * BM;

    const v8f vzero = {0.f,0.f,0.f,0.f,0.f,0.f,0.f,0.f};

    // ---- stage W2 hi/lo into LDS once (rows >=10 handled by zero fragments)
    for (int i = tid; i < DOUT * DH; i += 256) {
        const int   r  = i >> 9;          // 0..9
        const int   cc = i & 511;         // 0..511
        const float w  = W2[r * DH + cc];
        const __bf16 h = (__bf16)w;
        s_w2h[r * SPAD + cc] = h;
        s_w2l[r * SPAD + cc] = (__bf16)(w - (float)h);
    }
    // (visibility to waves 0/1 guaranteed by the first per-step __syncthreads)

    // ---------------- fc1: cur1[32x512] = x_tile @ W1^T (bf16 split x3) ----
    v8f cur1[2][4];
    #pragma unroll
    for (int mt = 0; mt < 2; ++mt)
        #pragma unroll
        for (int nt = 0; nt < 4; ++nt) cur1[mt][nt] = vzero;

    #pragma unroll 1
    for (int c = 0; c < 24; ++c) {        // branch-free main K loop
        FC1_CHUNK(c * 32, false);
    }
    FC1_CHUNK(768, true);                 // peeled tail (K 768..783 valid)

    // + b1 (column bias: every element a lane owns shares column wav*64+nt*16+lh)
    #pragma unroll
    for (int nt = 0; nt < 4; ++nt) {
        const float bias = b1[wav*64 + nt*16 + lh];
        #pragma unroll
        for (int mt = 0; mt < 2; ++mt)
            #pragma unroll
            for (int r = 0; r < 8; ++r) cur1[mt][nt][r] += bias;
    }

    // ---------------- LIF state -------------------------------------------
    v8f mem1[2][4];
    #pragma unroll
    for (int mt = 0; mt < 2; ++mt)
        #pragma unroll
        for (int nt = 0; nt < 4; ++nt) mem1[mt][nt] = vzero;

    v8f   mem2 = vzero;
    float b2v  = 0.f;
    if (wav < 2 && lh < DOUT) b2v = b2[lh];

    for (int s = 0; s < STEPS; ++s) {
        // layer-1 LIF update + spike store (bf16, exact 0/1) to LDS [row][k]
        #pragma unroll
        for (int mt = 0; mt < 2; ++mt)
            #pragma unroll
            for (int nt = 0; nt < 4; ++nt) {
                const int col = wav*64 + nt*16 + lh;
                #pragma unroll
                for (int r = 0; r < 8; ++r) {
                    float m   = mem1[mt][nt][r];
                    float rst = (m > 1.0f) ? 1.0f : 0.0f;     // reset from prev mem
                    m = 0.9f*m + cur1[mt][nt][r] - rst;
                    mem1[mt][nt][r] = m;
                    const int rowl = mt*16 + hh*8 + r;
                    s_spk[rowl*SPAD + col] = (m > 1.0f) ? (__bf16)1.0f : (__bf16)0.0f;
                }
            }
        __syncthreads();

        // layer-2: waves 0/1 each do one 16x16 tile (N padded 10->16), K=512
        if (wav < 2) {
            v8f acc2 = vzero;
            const int arow = wav*16 + lh;
            for (int c2 = 0; c2 < 16; ++c2) {
                const int kk = c2 * 32;
                // A fragment from LDS spikes
                const v8bf q0 = *(const v8bf*)&s_spk[arow*SPAD + kk + hh*8];
                const v8bf q1 = *(const v8bf*)&s_spk[arow*SPAD + kk + 16 + hh*8];
                v16bf a;
                #pragma unroll
                for (int i = 0; i < 8; ++i) { a[i] = q0[i]; a[8+i] = q1[i]; }
                // B fragment = staged W2 row lh (zero rows for padded N)
                v16bf bh, bl;
                if (lh < DOUT) {
                    const int wo = lh*SPAD + kk + hh*8;
                    const v8bf h0 = *(const v8bf*)&s_w2h[wo];
                    const v8bf h1 = *(const v8bf*)&s_w2h[wo + 16];
                    const v8bf l0 = *(const v8bf*)&s_w2l[wo];
                    const v8bf l1 = *(const v8bf*)&s_w2l[wo + 16];
                    #pragma unroll
                    for (int i = 0; i < 8; ++i) {
                        bh[i] = h0[i]; bh[8+i] = h1[i];
                        bl[i] = l0[i]; bl[8+i] = l1[i];
                    }
                } else {
                    #pragma unroll
                    for (int i = 0; i < 16; ++i) { bh[i] = (__bf16)0.f; bl[i] = (__bf16)0.f; }
                }
                acc2 = WMMA_BF16(a, bh, acc2);
                acc2 = WMMA_BF16(a, bl, acc2);
            }
            // layer-2 LIF + write spk_rec / mem_rec
            #pragma unroll
            for (int r = 0; r < 8; ++r) {
                const float c2v = acc2[r] + b2v;
                float m   = mem2[r];
                float rst = (m > 1.0f) ? 1.0f : 0.0f;
                m = 0.9f*m + c2v - rst;
                mem2[r] = m;
                if (lh < DOUT) {
                    const int    grow = mbase + wav*16 + hh*8 + r;
                    const size_t idx  = (size_t)s * (BATCH*DOUT) + (size_t)grow * DOUT + lh;
                    out[idx] = (m > 1.0f) ? 1.0f : 0.0f;
                    out[(size_t)STEPS*BATCH*DOUT + idx] = m;
                }
            }
        }
        __syncthreads();
    }
}

extern "C" void kernel_launch(void* const* d_in, const int* in_sizes, int n_in,
                              void* d_out, int out_size, void* d_ws, size_t ws_size,
                              hipStream_t stream) {
    const float* x  = (const float*)d_in[0];   // [32768, 784]
    const float* W1 = (const float*)d_in[1];   // [512, 784]
    const float* b1 = (const float*)d_in[2];   // [512]
    const float* W2 = (const float*)d_in[3];   // [10, 512]
    const float* b2 = (const float*)d_in[4];   // [10]
    float* out = (float*)d_out;                // [2 * 25 * 32768 * 10]

    dim3 grid(BATCH / BM);   // 1024 blocks
    dim3 block(256);         // 8 waves (wave32)
    hipLaunchKernelGGL(snn_fused, grid, block, 0, stream, x, W1, b1, W2, b2, out);
}
